// AttentionBlock_60842506715722
// MI455X (gfx1250) — matrix operand order repaired and checked by reading, not compiled, not run on hardware
//
#include <hip/hip_runtime.h>
#include <hip/hip_bf16.h>
#include <math.h>

typedef __attribute__((ext_vector_type(16))) _Float16 v16h;
typedef __attribute__((ext_vector_type(8)))  _Float16 v8h;
typedef __attribute__((ext_vector_type(4)))  _Float16 v4h;
typedef __attribute__((ext_vector_type(8)))  float    v8f;

#define TILE_M 256
#define TILE_N 128
#define TILE_K 32
#define LDS_PITCH 40   // halves per row: 32 data + 8 pad -> 80B rows, 16B aligned, bank-friendly

// ---------------------------------------------------------------------------
// f32 -> f16 cast, 4 elements/thread (b128 global load, b64 store)
// ---------------------------------------------------------------------------
__global__ __launch_bounds__(256) void cast_f32_f16(const float* __restrict__ src,
                                                    _Float16* __restrict__ dst, int n4) {
  int i = blockIdx.x * blockDim.x + threadIdx.x;
  if (i < n4) {
    float4 v = ((const float4*)src)[i];
    v4h h;
    h[0] = (_Float16)v.x; h[1] = (_Float16)v.y;
    h[2] = (_Float16)v.z; h[3] = (_Float16)v.w;
    ((v4h*)dst)[i] = h;
  }
}

// ---------------------------------------------------------------------------
// Fragment loads from LDS (wave32 layouts per CDNA5 ISA 7.12.2)
// A 16x32 f16: lane<16 -> M=lane, K={kb..kb+7, kb+16..kb+23}, kb = 0 (lo) / 8 (hi lanes)
// B 32x16 f16 staged as [N][K]: lane<16 -> N=lane, K={kb..kb+7, kb+16..kb+23}, kb = 0 (lo) / 8 (hi lanes)
// ---------------------------------------------------------------------------
__device__ __forceinline__ v16h load_frag_a(const _Float16* lds, int row0, int lane) {
  const int r  = row0 + (lane & 15);
  const int kb = (lane < 16) ? 0 : 8;
  const _Float16* p = lds + r * LDS_PITCH + kb;
  v8h lo = *(const v8h*)(p);        // K = kb .. kb+7
  v8h hi = *(const v8h*)(p + 16);   // K = kb+16 .. kb+23
  return __builtin_shufflevector(lo, hi, 0,1,2,3,4,5,6,7,8,9,10,11,12,13,14,15);
}

__device__ __forceinline__ v16h load_frag_b(const _Float16* lds, int col0, int lane) {
  const int r  = col0 + (lane & 15);
  const int ks = (lane < 16) ? 0 : 8;
  const _Float16* p = lds + r * LDS_PITCH + ks;
  v8h lo = *(const v8h*)(p);        // K = ks .. ks+7
  v8h hi = *(const v8h*)(p + 16);   // K = ks+16 .. ks+23
  return __builtin_shufflevector(lo, hi, 0,1,2,3,4,5,6,7,8,9,10,11,12,13,14,15);
}

// ---------------------------------------------------------------------------
// Stage one 256x32 A tile and one 128x32 B tile (as [N][K]) into LDS buffers.
// ---------------------------------------------------------------------------
template <bool B_NMAJOR>
__device__ __forceinline__ void stage_tiles(const _Float16* __restrict__ Ab,
                                            const _Float16* __restrict__ Bb,
                                            _Float16* __restrict__ la,
                                            _Float16* __restrict__ lb,
                                            int m0, int n0, int k0, int N, int Kd, int tid) {
  // A: one full 32-half row per thread (4x 16B loads/stores)
  {
    const int r = tid;  // 0..255
    const _Float16* g = Ab + (long long)(m0 + r) * Kd + k0;
    _Float16* d = la + r * LDS_PITCH;
    *(v8h*)(d)      = *(const v8h*)(g);
    *(v8h*)(d + 8)  = *(const v8h*)(g + 8);
    *(v8h*)(d + 16) = *(const v8h*)(g + 16);
    *(v8h*)(d + 24) = *(const v8h*)(g + 24);
    if (k0 + 2 * TILE_K < Kd) __builtin_prefetch(g + 2 * TILE_K, 0, 3);
  }
  if constexpr (B_NMAJOR) {
    // B stored [N][K]: same pattern as A, 2 threads per row, 128 rows
    const int r   = tid >> 1;
    const int seg = (tid & 1) * 16;
    const _Float16* g = Bb + (long long)(n0 + r) * Kd + k0 + seg;
    _Float16* d = lb + r * LDS_PITCH + seg;
    *(v8h*)(d)     = *(const v8h*)(g);
    *(v8h*)(d + 8) = *(const v8h*)(g + 8);
    if (k0 + 2 * TILE_K < Kd) __builtin_prefetch(g + 2 * TILE_K, 0, 3);
  } else {
    // B stored [K][N]: transpose while staging (16 scattered b16 stores)
    const int kk  = tid >> 3;         // 0..31  (K row)
    const int grp = (tid & 7) * 16;   // 0..112 (N chunk)
    const _Float16* g = Bb + (long long)(k0 + kk) * N + n0 + grp;
    v8h lo = *(const v8h*)(g);
    v8h hi = *(const v8h*)(g + 8);
#pragma unroll
    for (int j = 0; j < 8; ++j) {
      lb[(grp + j)     * LDS_PITCH + kk] = lo[j];
      lb[(grp + 8 + j) * LDS_PITCH + kk] = hi[j];
    }
    if (k0 + 2 * TILE_K < Kd) __builtin_prefetch(g + (long long)(2 * TILE_K) * N, 0, 3);
  }
}

// ---------------------------------------------------------------------------
// Tiled WMMA GEMM:  C[z] = op(A[z] x B[z])
//   Workgroup tile 256x128, 8 waves of 64x64 (4x4 WMMA frags each),
//   TILE_K=32, double-buffered LDS staging (one barrier per K step).
//   A: row-major MxK (lda = Kd)
//   B_NMAJOR=true : B stored [N][K] (ldb = Kd)  -> scores Q * K^T
//   B_NMAJOR=false: B stored [K][N] (ldb = N)   -> transposed while staging
// Epilogue: *scale, +bias[col], *mask[z*M+row], f16 or f32 store.
// M % 256 == 0, N % 128 == 0, Kd % 32 == 0 (guaranteed by problem sizes).
// ---------------------------------------------------------------------------
template <bool B_NMAJOR, bool OUT_HALF, bool HAS_BIAS, bool HAS_MASK>
__global__ __launch_bounds__(256)
void wmma_gemm(const _Float16* __restrict__ A, const _Float16* __restrict__ Bm,
               const float* __restrict__ bias, const float* __restrict__ mask,
               void* __restrict__ Cout, int M, int N, int Kd,
               long long strideA, long long strideB, long long strideC, float scale) {
  __shared__ _Float16 lds_a[2][TILE_M * LDS_PITCH];  // 2 x 20 KB
  __shared__ _Float16 lds_b[2][TILE_N * LDS_PITCH];  // 2 x 10 KB

  const int z = blockIdx.z;
  const _Float16* Ab = A  + (long long)z * strideA;
  const _Float16* Bb = Bm + (long long)z * strideB;

  const int n0 = blockIdx.x * TILE_N;
  const int m0 = blockIdx.y * TILE_M;

  const int tid  = threadIdx.x;
  const int lane = tid & 31;
  const int wave = tid >> 5;
  const int wm = (wave & 3) * 64;   // wave row offset within tile (4 waves down)
  const int wn = (wave >> 2) * 64;  // wave col offset within tile (2 waves across)

  v8f acc[4][4];
#pragma unroll
  for (int i = 0; i < 4; ++i)
#pragma unroll
    for (int j = 0; j < 4; ++j) acc[i][j] = {};

  stage_tiles<B_NMAJOR>(Ab, Bb, lds_a[0], lds_b[0], m0, n0, 0, N, Kd, tid);

  int buf = 0;
  for (int k0 = 0; k0 < Kd; k0 += TILE_K) {
    __syncthreads();  // staged tile `buf` visible; prev compute on `buf^1` done
    if (k0 + TILE_K < Kd)
      stage_tiles<B_NMAJOR>(Ab, Bb, lds_a[buf ^ 1], lds_b[buf ^ 1],
                            m0, n0, k0 + TILE_K, N, Kd, tid);

    v16h af[4], bf[4];
#pragma unroll
    for (int i = 0; i < 4; ++i) af[i] = load_frag_a(lds_a[buf], wm + i * 16, lane);
#pragma unroll
    for (int j = 0; j < 4; ++j) bf[j] = load_frag_b(lds_b[buf], wn + j * 16, lane);
#pragma unroll
    for (int i = 0; i < 4; ++i)
#pragma unroll
      for (int j = 0; j < 4; ++j)
        acc[i][j] = __builtin_amdgcn_wmma_f32_16x16x32_f16(
            false, af[i], false, bf[j], (short)0, acc[i][j], false, false);
    buf ^= 1;
  }

  // ---- epilogue: C/D layout -> lane<16: N=lane, M=+e ; lane>=16: M=+8+e ----
  const int colL  = lane & 15;
  const int rbase = (lane < 16) ? 0 : 8;
  _Float16* Ch = (_Float16*)Cout + (long long)z * strideC;
  float*    Cf = (float*)Cout    + (long long)z * strideC;
#pragma unroll
  for (int i = 0; i < 4; ++i) {
#pragma unroll
    for (int j = 0; j < 4; ++j) {
      const int col = n0 + wn + j * 16 + colL;
      const float bb = HAS_BIAS ? bias[col] : 0.0f;
#pragma unroll
      for (int e = 0; e < 8; ++e) {
        const int row = m0 + wm + i * 16 + rbase + e;
        float v = acc[i][j][e] * scale + bb;
        if constexpr (HAS_MASK) v *= mask[(long long)z * M + row];
        if constexpr (OUT_HALF) Ch[(long long)row * N + col] = (_Float16)v;
        else                    Cf[(long long)row * N + col] = v;
      }
    }
  }
}

// ---------------------------------------------------------------------------
// Softmax over the QUERY axis (axis=1): column softmax of S[b] (T x T, f16).
// One thread per (b, column); coalesced row sweeps; online max/sum in f32.
// ---------------------------------------------------------------------------
__global__ __launch_bounds__(256) void softmax_col(_Float16* __restrict__ S, int T) {
  const int b   = blockIdx.y;
  const int col = blockIdx.x * blockDim.x + threadIdx.x;
  _Float16* Sb = S + (long long)b * T * T;

  float m = -INFINITY, l = 0.0f;
  for (int q = 0; q < T; ++q) {
    const float s  = (float)Sb[(long long)q * T + col];
    const float mn = fmaxf(m, s);
    l = l * __expf(m - mn) + __expf(s - mn);
    m = mn;
  }
  const float inv = 1.0f / l;
  for (int q = 0; q < T; ++q) {
    const long long idx = (long long)q * T + col;
    const float s = (float)Sb[idx];
    Sb[idx] = (_Float16)(__expf(s - m) * inv);
  }
}

// ---------------------------------------------------------------------------
// Host-side orchestration
// ---------------------------------------------------------------------------
extern "C" void kernel_launch(void* const* d_in, const int* in_sizes, int n_in,
                              void* d_out, int out_size, void* d_ws, size_t ws_size,
                              hipStream_t stream) {
  (void)in_sizes; (void)n_in; (void)out_size; (void)ws_size;
  const float* x_f  = (const float*)d_in[0];
  const float* mask = (const float*)d_in[1];
  const float* Wq   = (const float*)d_in[2];
  const float* bq   = (const float*)d_in[3];
  const float* Wk   = (const float*)d_in[4];
  const float* bk   = (const float*)d_in[5];
  const float* Wv   = (const float*)d_in[6];
  const float* bv   = (const float*)d_in[7];
  float* out = (float*)d_out;

  constexpr int  Bn = 8, T = 2048, C = 1024, KS = 1024, VS = 1024;
  constexpr long long XN = (long long)Bn * T * C;  // 16,777,216 halves
  constexpr long long WN = (long long)C * KS;      //  1,048,576 halves

  // Workspace layout (f16 everywhere): ~198 MB total
  _Float16* x16 = (_Float16*)d_ws;
  _Float16* w16 = x16 + XN;        // 3 weight matrices, contiguous
  _Float16* q16 = w16 + 3 * WN;
  _Float16* k16 = q16 + XN;
  _Float16* v16 = k16 + XN;
  _Float16* s16 = v16 + XN;        // scores / attn, B*T*T halves

  // 1) casts to f16
  cast_f32_f16<<<(int)(XN / 4 / 256), 256, 0, stream>>>(x_f, x16, (int)(XN / 4));
  cast_f32_f16<<<(int)(WN / 4 / 256), 256, 0, stream>>>(Wq, w16,          (int)(WN / 4));
  cast_f32_f16<<<(int)(WN / 4 / 256), 256, 0, stream>>>(Wk, w16 + WN,     (int)(WN / 4));
  cast_f32_f16<<<(int)(WN / 4 / 256), 256, 0, stream>>>(Wv, w16 + 2 * WN, (int)(WN / 4));

  // 2) QKV projections: [16384 x 1024] * [1024 x 1024] + bias -> f16
  {
    dim3 g(KS / TILE_N, (Bn * T) / TILE_M, 1);
    wmma_gemm<false, true, true, false><<<g, 256, 0, stream>>>(
        x16, w16,          bq, nullptr, q16, Bn * T, KS, C, 0, 0, 0, 1.0f);
    wmma_gemm<false, true, true, false><<<g, 256, 0, stream>>>(
        x16, w16 + WN,     bk, nullptr, k16, Bn * T, KS, C, 0, 0, 0, 1.0f);
    wmma_gemm<false, true, true, false><<<g, 256, 0, stream>>>(
        x16, w16 + 2 * WN, bv, nullptr, v16, Bn * T, VS, C, 0, 0, 0, 1.0f);
  }

  // 3) scores[b] = Q[b] * K[b]^T / sqrt(1024)   (B stored [N][K] -> B_NMAJOR)
  {
    dim3 g(T / TILE_N, T / TILE_M, Bn);
    wmma_gemm<true, true, false, false><<<g, 256, 0, stream>>>(
        q16, k16, nullptr, nullptr, s16, T, T, KS,
        (long long)T * KS, (long long)T * KS, (long long)T * T, 1.0f / 32.0f);
  }

  // 4) softmax over query axis (columns of S[b]); attn written in place (f16)
  softmax_col<<<dim3(T / 256, Bn), 256, 0, stream>>>(s16, T);

  // 5) out[b] = attn[b] * V[b], f32 output, * mask[b, row]
  {
    dim3 g(VS / TILE_N, T / TILE_M, Bn);
    wmma_gemm<false, false, false, true><<<g, 256, 0, stream>>>(
        s16, v16, nullptr, mask, out, T, VS, T,
        (long long)T * T, (long long)T * VS, (long long)T * VS, 1.0f);
  }
}
